// DecoderRNN_90683939488321
// MI455X (gfx1250) — compile-verified, matrix operand order checked by reading, not run on hardware
//
#include <hip/hip_runtime.h>
#include <hip/hip_bf16.h>
#include <stddef.h>
#include <stdint.h>

// ---------------------------------------------------------------------------
// DA-RNN decoder for MI455X (gfx1250, wave32, WMMA + async-to-LDS)
// B=512, Tm1=255, EH=DH=128, OF=1
// ---------------------------------------------------------------------------

#define B_    512
#define TM1   255
#define EH_   128
#define DH_   128
#define LDK   136   // padded f16 row stride for K=128 (272 B: 16B-aligned, bank-skewed)
#define LDK2  264   // padded f16 row stride for K=256 (528 B: 16B-aligned, bank-skewed)
#define GPAD  520   // padded f32 gate row stride

typedef __attribute__((ext_vector_type(16))) _Float16 v16h;
typedef __attribute__((ext_vector_type(8)))  _Float16 v8h;
typedef __attribute__((ext_vector_type(8)))  float    v8f;

// Load one 16x32 (f16) WMMA operand fragment for this lane from a row-major
// [16][K] block. `p` points at (row, kchunk) base; khalf = lane>>4 selects
// which K-half this lane holds. Two 16-byte loads per fragment.
__device__ __forceinline__ v16h load_frag(const _Float16* p, int khalf) {
  v8h lo = *(const v8h*)(p + 8 * khalf);
  v8h hi = *(const v8h*)(p + 16 + 8 * khalf);
  v16h r;
#pragma unroll
  for (int i = 0; i < 8; ++i) { r[i] = lo[i]; r[8 + i] = hi[i]; }
  return r;
}

// Async copy 16 bytes global -> LDS (gfx1250 GLOBAL_LOAD_ASYNC_TO_LDS_B128,
// tracked by ASYNCcnt). LDS byte offset = low 32 bits of generic shared ptr
// (flat shared-aperture mapping).
__device__ __forceinline__ void async_copy_b128(void* lds_dst, const void* gsrc) {
  uint32_t lds = (uint32_t)(uintptr_t)lds_dst;
  uint64_t ga  = (uint64_t)(uintptr_t)gsrc;
  asm volatile("global_load_async_to_lds_b128 %0, %1, off"
               :: "v"(lds), "v"(ga) : "memory");
}
__device__ __forceinline__ void wait_async_zero() {
  asm volatile("s_wait_asynccnt 0x0" ::: "memory");
}

__device__ __forceinline__ float sigf(float x) { return 1.0f / (1.0f + __expf(-x)); }

// ---------------------------------------------------------------------------
// prep: f32 -> f16 weight conversion
//   w1enc = attn_W1[:, 256:384]  (128 x 128)
//   w1hc  = attn_W1[:, 0:256]    (128 x 256)
//   whh   = lstm_W_hh            (512 x 128)
// ---------------------------------------------------------------------------
__global__ void __launch_bounds__(256) prep_kernel(const float* __restrict__ attn_W1,
                                                   const float* __restrict__ lstm_W_hh,
                                                   _Float16* __restrict__ w1enc,
                                                   _Float16* __restrict__ w1hc,
                                                   _Float16* __restrict__ whh) {
  int i = blockIdx.x * 256 + threadIdx.x;
  if (i < EH_ * EH_) {
    int h = i >> 7, e = i & 127;
    w1enc[i] = (_Float16)attn_W1[h * 384 + 256 + e];
  }
  if (i < EH_ * 256) {
    int h = i >> 8, j = i & 255;
    w1hc[i] = (_Float16)attn_W1[h * 384 + j];
  }
  if (i < 4 * DH_ * DH_) {
    whh[i] = (_Float16)lstm_W_hh[i];
  }
}

// ---------------------------------------------------------------------------
// zero: initialize h, c, context, y_tilde state (contiguous block)
// ---------------------------------------------------------------------------
__global__ void __launch_bounds__(256) zero_kernel(float* __restrict__ p, int n) {
  for (int i = blockIdx.x * 256 + threadIdx.x; i < n; i += gridDim.x * 256) p[i] = 0.0f;
}

// ---------------------------------------------------------------------------
// encproj: enc_proj[m][h] = sum_e input[m][e] * W1_enc[h][e]
// M = B*TM1 = 130560 rows; one WG (8 waves) per 16-row M tile; wave w does
// the 16x16 tile at N = 16*w. K = 128 in 4 chunks of 32 via WMMA f16->f32.
// W1_enc is staged into LDS with async-to-LDS b128 copies (ASYNCcnt).
// ---------------------------------------------------------------------------
__global__ void __launch_bounds__(256) encproj_kernel(const float* __restrict__ inp,
                                                      const _Float16* __restrict__ w1enc,
                                                      float* __restrict__ encp) {
  __shared__ _Float16 sA[16 * LDK];      //  4.25 KB
  __shared__ _Float16 sB[128 * LDK];     // 34.0  KB  (whole W1_enc)
  const int mt = blockIdx.x;             // 0..8159
  const int tid = threadIdx.x, lane = tid & 31, wave = tid >> 5;

  // stage B = W1_enc (128 rows x 128 f16) via async b128 copies: 2048 chunks
#pragma unroll
  for (int ci = 0; ci < 8; ++ci) {
    int i = tid + ci * 256;              // chunk index
    int n = i >> 4, c = i & 15;          // row, 8-half chunk in row
    async_copy_b128(&sB[n * LDK + c * 8], w1enc + i * 8);
  }
  // stage A tile (16 x 128), converting f32 -> f16 (VALU path)
  for (int i = tid; i < 16 * 128; i += 256) {
    int r = i >> 7, k = i & 127;
    sA[r * LDK + k] = (_Float16)inp[((size_t)mt * 16 + r) * 128 + k];
  }
  wait_async_zero();
  __syncthreads();

  const int row = lane & 15, khalf = lane >> 4;
  const int n0 = wave * 16;
  v8f acc = {};
#pragma unroll
  for (int kc = 0; kc < 4; ++kc) {
    v16h a = load_frag(&sA[row * LDK + kc * 32], khalf);
    v16h b = load_frag(&sB[(n0 + row) * LDK + kc * 32], khalf);
    acc = __builtin_amdgcn_wmma_f32_16x16x32_f16(false, a, false, b, (short)0, acc,
                                                 false, false);
  }
  // D layout: lane L -> col L%16, rows v + 8*(L/16)
  const int col = n0 + (lane & 15);
  const int rb = 8 * (lane >> 4);
#pragma unroll
  for (int v = 0; v < 8; ++v)
    encp[((size_t)mt * 16 + rb + v) * 128 + col] = acc[v];
}

// ---------------------------------------------------------------------------
// hcproj: HP = [H|C](512x256) @ W1_hc^T(256x128) + b1, per step, via WMMA.
// 32 WGs x 16 batch rows; 8 waves each own one 16x16 N tile (N=128 total).
// ---------------------------------------------------------------------------
__global__ void __launch_bounds__(256) hcproj_kernel(const float* __restrict__ hbuf,
                                                     const float* __restrict__ cbuf,
                                                     const _Float16* __restrict__ w1hc,
                                                     const float* __restrict__ attn_b1,
                                                     float* __restrict__ hp) {
  __shared__ _Float16 sA[16 * LDK2];     // 8.25 KB, [row][k] k<256 = concat(h,c)
  const int mt = blockIdx.x;             // 0..31
  const int tid = threadIdx.x, lane = tid & 31, wave = tid >> 5;

  for (int i = tid; i < 16 * 256; i += 256) {
    int r = i >> 8, k = i & 255;
    float v = (k < 128) ? hbuf[(mt * 16 + r) * 128 + k]
                        : cbuf[(mt * 16 + r) * 128 + (k - 128)];
    sA[r * LDK2 + k] = (_Float16)v;
  }
  __syncthreads();

  const int row = lane & 15, khalf = lane >> 4;
  const int n0 = wave * 16;
  const _Float16* aBase = sA + row * LDK2;
  const _Float16* bRow = w1hc + (size_t)(n0 + row) * 256;
  v8f acc = {};
#pragma unroll
  for (int kc = 0; kc < 8; ++kc) {
    v16h a = load_frag(aBase + kc * 32, khalf);
    v16h b = load_frag(bRow + kc * 32, khalf);
    acc = __builtin_amdgcn_wmma_f32_16x16x32_f16(false, a, false, b, (short)0, acc,
                                                 false, false);
  }
  const int col = n0 + (lane & 15);
  const int rb = 8 * (lane >> 4);
  const float b1 = attn_b1[col];
#pragma unroll
  for (int v = 0; v < 8; ++v)
    hp[(mt * 16 + rb + v) * 128 + col] = acc[v] + b1;
}

// ---------------------------------------------------------------------------
// attn_step: one WG per batch row b.
//   score[t] = w2 . tanh(hp[b] + enc_proj[b,t,:]) + b2
//   attn = softmax_t(score);  context = sum_t attn[t] * input[b,t,:]
//   y_tilde = fc_W[:128].context + fc_W[128]*y_t + fc_b
// ---------------------------------------------------------------------------
__global__ void __launch_bounds__(256) attn_step_kernel(
    const float* __restrict__ inp, const float* __restrict__ encp,
    const float* __restrict__ hp_g, const float* __restrict__ attn_W2,
    const float* __restrict__ attn_b2, const float* __restrict__ y_history,
    const float* __restrict__ fc_W, const float* __restrict__ fc_b,
    float* __restrict__ ctxbuf, float* __restrict__ ytilde, int step) {
  __shared__ float s_hp[128];
  __shared__ float s_sc[256];
  __shared__ float s_red[256];

  const int b = blockIdx.x;
  const int tid = threadIdx.x, lane = tid & 31, wave = tid >> 5;

  if (tid < 128) s_hp[tid] = hp_g[b * 128 + tid];  // bias already folded in
  __syncthreads();

  const float4 hp = *(const float4*)(s_hp + 4 * lane);
  const float4 w2 = *(const float4*)(attn_W2 + 4 * lane);
  const float b2 = attn_b2[0];
  const float* ep_b = encp + (size_t)b * TM1 * 128;

  for (int t = wave; t < TM1; t += 8) {
    if (t + 8 < TM1) __builtin_prefetch(ep_b + (t + 8) * 128 + 4 * lane, 0, 1);
    const float4 e = *(const float4*)(ep_b + t * 128 + 4 * lane);
    float s = w2.x * tanhf(hp.x + e.x) + w2.y * tanhf(hp.y + e.y) +
              w2.z * tanhf(hp.z + e.z) + w2.w * tanhf(hp.w + e.w);
#pragma unroll
    for (int m = 16; m > 0; m >>= 1) s += __shfl_xor(s, m, 32);
    if (lane == 0) s_sc[t] = s + b2;
  }
  __syncthreads();

  // softmax over 255 scores
  const float sv = (tid < TM1) ? s_sc[tid] : -3.4e38f;
  s_red[tid] = sv;
  __syncthreads();
#pragma unroll
  for (int off = 128; off > 0; off >>= 1) {
    if (tid < off) s_red[tid] = fmaxf(s_red[tid], s_red[tid + off]);
    __syncthreads();
  }
  const float mx = s_red[0];
  __syncthreads();
  const float ev = (tid < TM1) ? __expf(sv - mx) : 0.0f;
  if (tid < TM1) s_sc[tid] = ev;
  s_red[tid] = ev;
  __syncthreads();
#pragma unroll
  for (int off = 128; off > 0; off >>= 1) {
    if (tid < off) s_red[tid] += s_red[tid + off];
    __syncthreads();
  }
  const float inv = 1.0f / s_red[0];
  __syncthreads();

  // context: split t-range across the two 128-thread halves
  const int e = tid & 127, halfsel = tid >> 7;
  const float* ib = inp + (size_t)b * TM1 * 128;
  float acc = 0.0f;
  const int t0 = halfsel * 128, t1 = halfsel ? TM1 : 128;
  for (int t = t0; t < t1; ++t) acc += s_sc[t] * ib[t * 128 + e];
  s_red[tid] = acc;
  __syncthreads();

  float ctxv = 0.0f;
  if (tid < 128) {
    ctxv = (s_red[tid] + s_red[tid + 128]) * inv;
    ctxbuf[b * 128 + tid] = ctxv;
  }
  __syncthreads();

  // y_tilde
  s_red[tid] = (tid < 128) ? fc_W[tid] * ctxv : 0.0f;
  __syncthreads();
#pragma unroll
  for (int off = 128; off > 0; off >>= 1) {
    if (tid < off) s_red[tid] += s_red[tid + off];
    __syncthreads();
  }
  if (tid == 0)
    ytilde[b] = s_red[0] + fc_W[128] * y_history[(size_t)b * TM1 + step] + fc_b[0];
}

// ---------------------------------------------------------------------------
// lstm_step: gates = h @ W_hh^T (+ y_tilde*W_ih + biases), LSTM update.
// 32 WGs x 16 batch rows; 8 waves x 4 passes of 16x16 WMMA tiles cover N=512.
// ---------------------------------------------------------------------------
__global__ void __launch_bounds__(256) lstm_step_kernel(
    const float* __restrict__ ytilde, const _Float16* __restrict__ whh,
    const float* __restrict__ W_ih, const float* __restrict__ b_ih,
    const float* __restrict__ b_hh, float* __restrict__ hbuf,
    float* __restrict__ cbuf) {
  __shared__ _Float16 sA[16 * LDK];   //  4.25 KB
  __shared__ float sG[16 * GPAD];     // 32.5  KB
  const int wg = blockIdx.x;          // 0..31
  const int tid = threadIdx.x, lane = tid & 31, wave = tid >> 5;

  for (int i = tid; i < 16 * 128; i += 256) {
    int r = i >> 7, k = i & 127;
    sA[r * LDK + k] = (_Float16)hbuf[(wg * 16 + r) * 128 + k];
  }
  __syncthreads();

  const int row = lane & 15, khalf = lane >> 4;
  const _Float16* aBase = sA + row * LDK;
#pragma unroll
  for (int pass = 0; pass < 4; ++pass) {
    const int n0 = pass * 128 + wave * 16;
    const _Float16* bRow = whh + (size_t)(n0 + row) * 128;
    v8f acc = {};
#pragma unroll
    for (int kc = 0; kc < 4; ++kc) {
      v16h a = load_frag(aBase + kc * 32, khalf);
      v16h b = load_frag(bRow + kc * 32, khalf);
      acc = __builtin_amdgcn_wmma_f32_16x16x32_f16(false, a, false, b, (short)0, acc,
                                                   false, false);
    }
    const int col = n0 + (lane & 15);
    const int rb = 8 * (lane >> 4);
#pragma unroll
    for (int v = 0; v < 8; ++v) sG[(rb + v) * GPAD + col] = acc[v];
  }
  __syncthreads();

  for (int i = tid; i < 16 * 128; i += 256) {
    const int r = i >> 7, d = i & 127;
    const int b = wg * 16 + r;
    const float yt = ytilde[b];
    const float gi = sG[r * GPAD + d] + yt * W_ih[d] + b_ih[d] + b_hh[d];
    const float gf = sG[r * GPAD + 128 + d] + yt * W_ih[128 + d] + b_ih[128 + d] + b_hh[128 + d];
    const float gg = sG[r * GPAD + 256 + d] + yt * W_ih[256 + d] + b_ih[256 + d] + b_hh[256 + d];
    const float go = sG[r * GPAD + 384 + d] + yt * W_ih[384 + d] + b_ih[384 + d] + b_hh[384 + d];
    const float cn = sigf(gf) * cbuf[b * 128 + d] + sigf(gi) * tanhf(gg);
    hbuf[b * 128 + d] = sigf(go) * tanhf(cn);
    cbuf[b * 128 + d] = cn;
  }
}

// ---------------------------------------------------------------------------
// finout: out[b] = fcf_W[:128].h[b] + fcf_W[128:].context[b] + fcf_b
// ---------------------------------------------------------------------------
__global__ void __launch_bounds__(128) finout_kernel(const float* __restrict__ hbuf,
                                                     const float* __restrict__ ctxbuf,
                                                     const float* __restrict__ fcf_W,
                                                     const float* __restrict__ fcf_b,
                                                     float* __restrict__ out) {
  __shared__ float r[128];
  const int b = blockIdx.x, tid = threadIdx.x;
  r[tid] = fcf_W[tid] * hbuf[b * 128 + tid] + fcf_W[128 + tid] * ctxbuf[b * 128 + tid];
  __syncthreads();
#pragma unroll
  for (int off = 64; off > 0; off >>= 1) {
    if (tid < off) r[tid] += r[tid + off];
    __syncthreads();
  }
  if (tid == 0) out[b] = r[0] + fcf_b[0];
}

// ---------------------------------------------------------------------------
extern "C" void kernel_launch(void* const* d_in, const int* in_sizes, int n_in,
                              void* d_out, int out_size, void* d_ws, size_t ws_size,
                              hipStream_t stream) {
  const float* input_encoded = (const float*)d_in[0];
  const float* y_history     = (const float*)d_in[1];
  const float* attn_W1       = (const float*)d_in[2];
  const float* attn_b1       = (const float*)d_in[3];
  const float* attn_W2       = (const float*)d_in[4];
  const float* attn_b2       = (const float*)d_in[5];
  const float* lstm_W_ih     = (const float*)d_in[6];
  const float* lstm_W_hh     = (const float*)d_in[7];
  const float* lstm_b_ih     = (const float*)d_in[8];
  const float* lstm_b_hh     = (const float*)d_in[9];
  const float* fc_W          = (const float*)d_in[10];
  const float* fc_b          = (const float*)d_in[11];
  const float* fcf_W         = (const float*)d_in[12];
  const float* fcf_b         = (const float*)d_in[13];
  (void)in_sizes; (void)n_in; (void)out_size; (void)ws_size;

  char* ws = (char*)d_ws;
  size_t off = 0;
  float* encp = (float*)(ws + off);        off += (size_t)B_ * TM1 * EH_ * sizeof(float); // 66.8 MB
  _Float16* w1enc = (_Float16*)(ws + off); off += (size_t)EH_ * EH_ * sizeof(_Float16);
  _Float16* w1hc  = (_Float16*)(ws + off); off += (size_t)EH_ * 256 * sizeof(_Float16);
  _Float16* whh   = (_Float16*)(ws + off); off += (size_t)4 * DH_ * DH_ * sizeof(_Float16);
  float* hbuf   = (float*)(ws + off);      off += (size_t)B_ * DH_ * sizeof(float);
  float* cbuf   = (float*)(ws + off);      off += (size_t)B_ * DH_ * sizeof(float);
  float* ctxbuf = (float*)(ws + off);      off += (size_t)B_ * EH_ * sizeof(float);
  float* ytilde = (float*)(ws + off);      off += (size_t)B_ * sizeof(float);
  float* hp     = (float*)(ws + off);      off += (size_t)B_ * EH_ * sizeof(float);

  prep_kernel<<<256, 256, 0, stream>>>(attn_W1, lstm_W_hh, w1enc, w1hc, whh);
  zero_kernel<<<512, 256, 0, stream>>>(hbuf, B_ * DH_ * 2 + B_ * EH_ + B_); // h,c,ctx,ytilde
  encproj_kernel<<<(B_ * TM1) / 16, 256, 0, stream>>>(input_encoded, w1enc, encp);

  for (int s = 0; s < TM1; ++s) {
    hcproj_kernel<<<B_ / 16, 256, 0, stream>>>(hbuf, cbuf, w1hc, attn_b1, hp);
    attn_step_kernel<<<B_, 256, 0, stream>>>(input_encoded, encp, hp, attn_W2, attn_b2,
                                             y_history, fc_W, fc_b, ctxbuf, ytilde, s);
    lstm_step_kernel<<<B_ / 16, 256, 0, stream>>>(ytilde, whh, lstm_W_ih, lstm_b_ih,
                                                  lstm_b_hh, hbuf, cbuf);
  }

  finout_kernel<<<B_, 128, 0, stream>>>(hbuf, ctxbuf, fcf_W, fcf_b, (float*)d_out);
}
